// InteractionGate_43602507989723
// MI455X (gfx1250) — matrix operand order repaired and checked by reading, not compiled, not run on hardware
//
#include <hip/hip_runtime.h>
#include <hip/hip_bf16.h>
#include <math.h>

typedef __attribute__((ext_vector_type(16))) _Float16 v16h;
typedef __attribute__((ext_vector_type(8)))  float    v8f;
typedef __attribute__((ext_vector_type(4)))  float    f32x4;   // native vec for NT store

#define N_PTS 1024
#define HDIM  64
#define EDIM  16

// ---------------------------------------------------------------------------
// Kernel 0: tiny per-feature vectors
//   u_self[h]  = sum_e w_dist[e] * Wg[(128+e)*H + h]
//   u_other[h] = sum_e w_dist[e] * Wg[(144+e)*H + h]
//   cvec[h]    = b_gate[h] + sum_e b_dist[e]*(Wg[(128+e)*H+h] + Wg[(144+e)*H+h])
// ---------------------------------------------------------------------------
__global__ void ig_precompute_vecs(const float* __restrict__ wd,
                                   const float* __restrict__ bd,
                                   const float* __restrict__ Wg,
                                   const float* __restrict__ bg,
                                   float* __restrict__ us,
                                   float* __restrict__ uo,
                                   float* __restrict__ cv) {
    int h = threadIdx.x;  // 0..63
    float s = 0.f, o = 0.f, c = bg[h];
#pragma unroll
    for (int e = 0; e < EDIM; ++e) {
        float w2 = Wg[(2*HDIM + e) * HDIM + h];          // rows 128..143
        float w3 = Wg[(2*HDIM + EDIM + e) * HDIM + h];   // rows 144..159
        s += wd[e] * w2;
        o += wd[e] * w3;
        c += bd[e] * (w2 + w3);
    }
    us[h] = s;
    uo[h] = o;
    cv[h] = c;
}

// ---------------------------------------------------------------------------
// Kernel 1: WMMA precompute GEMMs (wave32, one 16x16 C tile per wave)
//   Acomb = Xa @ (W0 + W1)            [1024,64]
//   Bdiag = Xa @ W0 + Xg @ W1         [1024,64]
// f16 inputs, f32 accumulation via v_wmma_f32_16x16x32_f16 (K=64 -> 2 steps).
// ---------------------------------------------------------------------------
__global__ void ig_precompute_gemm(const float* __restrict__ Xa,
                                   const float* __restrict__ Xg,
                                   const float* __restrict__ Wg,
                                   float* __restrict__ Acomb,
                                   float* __restrict__ Bdiag) {
    const int wave = threadIdx.x >> 5;
    const int lane = threadIdx.x & 31;
    const int tile = blockIdx.x * 4 + wave;   // 256 tiles total (64 m x 4 n)
    const int m0 = (tile >> 2) * 16;
    const int n0 = (tile & 3) * 16;
    const int half = lane >> 4;               // 0 or 1 (lane group)
    const int mrow = m0 + (lane & 15);        // A-matrix row (M = lane%16)
    const int ncol = n0 + (lane & 15);        // B/C/D column (N = lane%16)

    v16h aa0, aa1, ag0, ag1;    // A fragments: Xa/Xg, K-steps 0 and 32
    v16h b00, b01, b10, b11;    // B fragments: W0/W1, K-steps 0 and 32
#pragma unroll
    for (int idx = 0; idx < 16; ++idx) {
        // 16-bit A 16x32 layout: K = idx + 8*half + (idx>=8 ? 8 : 0)
        int ka = idx + half * 8 + ((idx >= 8) ? 8 : 0);
        aa0[idx] = (_Float16)Xa[mrow * HDIM + ka];
        aa1[idx] = (_Float16)Xa[mrow * HDIM + 32 + ka];
        ag0[idx] = (_Float16)Xg[mrow * HDIM + ka];
        ag1[idx] = (_Float16)Xg[mrow * HDIM + 32 + ka];
        // 16-bit B 32x16 layout: K = idx + 8*half + (idx>=8 ? 8 : 0), N = lane%16
        int kb = idx + half * 8 + ((idx >= 8) ? 8 : 0);
        b00[idx] = (_Float16)Wg[(kb)        * HDIM + ncol];      // W0 rows  0..31
        b01[idx] = (_Float16)Wg[(32 + kb)   * HDIM + ncol];      // W0 rows 32..63
        b10[idx] = (_Float16)Wg[(HDIM + kb)      * HDIM + ncol]; // W1 rows  0..31
        b11[idx] = (_Float16)Wg[(HDIM + 32 + kb) * HDIM + ncol]; // W1 rows 32..63
    }

    v8f c1 = {};  // Xa@(W0+W1)
    v8f c2 = {};  // Xa@W0 + Xg@W1
    c1 = __builtin_amdgcn_wmma_f32_16x16x32_f16(false, aa0, false, b00, (short)0, c1, false, false);
    c1 = __builtin_amdgcn_wmma_f32_16x16x32_f16(false, aa1, false, b01, (short)0, c1, false, false);
    c1 = __builtin_amdgcn_wmma_f32_16x16x32_f16(false, aa0, false, b10, (short)0, c1, false, false);
    c1 = __builtin_amdgcn_wmma_f32_16x16x32_f16(false, aa1, false, b11, (short)0, c1, false, false);

    c2 = __builtin_amdgcn_wmma_f32_16x16x32_f16(false, aa0, false, b00, (short)0, c2, false, false);
    c2 = __builtin_amdgcn_wmma_f32_16x16x32_f16(false, aa1, false, b01, (short)0, c2, false, false);
    c2 = __builtin_amdgcn_wmma_f32_16x16x32_f16(false, ag0, false, b10, (short)0, c2, false, false);
    c2 = __builtin_amdgcn_wmma_f32_16x16x32_f16(false, ag1, false, b11, (short)0, c2, false, false);

    // 32-bit C/D 16x16 layout: VGPR r -> M = r + 8*half, N = lane%16
#pragma unroll
    for (int r = 0; r < 8; ++r) {
        int m = m0 + r + half * 8;
        Acomb[m * HDIM + ncol] = c1[r];
        Bdiag[m * HDIM + ncol] = c2[r];
    }
}

// ---------------------------------------------------------------------------
// Kernel 2: main store-bound kernel. One block = one i x 64 j's.
// Stage 1: 64 threads compute d_other[i, j0..j0+63] into LDS.
// Stage 2: float4 coalesced NON-TEMPORAL stores of goal_action*sigmoid(logit).
//          NT keeps the 256 MB write-once stream from evicting the ~1 MB
//          L2-resident read set (Acomb/Bdiag/hidden rows reused across all i).
// ---------------------------------------------------------------------------
__global__ void __launch_bounds__(256)
ig_main(const float* __restrict__ Ah,      // action_hidden [N,H]
        const float* __restrict__ Gh,      // goal_hidden   [N,H]
        const float* __restrict__ goal,    // [N,2]
        const float* __restrict__ action,  // [N,2]
        const float* __restrict__ Acomb,   // [N,H]
        const float* __restrict__ Bdiag,   // [N,H]
        const float* __restrict__ us,
        const float* __restrict__ uo,
        const float* __restrict__ cv,
        float* __restrict__ out) {
    __shared__ float sd[64];

    const int i   = blockIdx.y;
    const int j0  = blockIdx.x * 64;
    const int tid = threadIdx.x;

    // ds[i] = ||action_i - goal_i|| (tiny arrays, L2 resident)
    const float aix = action[2 * i], aiy = action[2 * i + 1];
    const float gix = goal[2 * i],   giy = goal[2 * i + 1];
    const float dsi = sqrtf((aix - gix) * (aix - gix) + (aiy - giy) * (aiy - giy));

    if (tid < 64) {
        int j = j0 + tid;
        float d;
        if (j == i) {
            d = dsi;  // diagonal: diagv[i]
        } else {
            // j>i: pair (i,j), take d1.  j<i: pair (j,i), take d2.
            int a = (j > i) ? i : j;
            int b = (j > i) ? j : i;
            float ax = action[2 * a], ay = action[2 * a + 1];
            float bx = action[2 * b], by = action[2 * b + 1];
            float gax = goal[2 * a], gay = goal[2 * a + 1];
            float gbx = goal[2 * b], gby = goal[2 * b + 1];
            // faithful to source, typos included
            float dn      = -(ay - by) * (gax - gbx);           // (ax-ax)*... == 0
            float cross_a = ax * by - ay * bx;
            float cross_g = gax * gby - gax * gbx;              // gx_i*gy_j - gx_i*gx_j
            float n1 = cross_a * (gax - gbx) - (ax - bx) * cross_g;
            float n2 = cross_a * (gay - gby) - (ay - bx) * cross_g;  // (ay_i - ax_j)
            float safe = (dn == 0.f) ? 1.f : dn;
            float p1 = n1 / safe, p2 = n2 / safe;
            bool cond = (dn != 0.f) && ((ax - p1) * (p1 - gax) > 0.f);
            float dp;
            if (j > i) dp = sqrtf((ax - p1) * (ax - p1) + (ay - p2) * (ay - p2)); // d1p
            else       dp = sqrtf((bx - p1) * (bx - p1) + (by - p2) * (by - p2)); // d2p
            d = cond ? dp : dsi;  // fallback is diagv[i] in both triangles
        }
        sd[tid] = d;
    }
    __syncthreads();

    const int h  = (tid & 15) * 4;   // 4 consecutive features per thread
    const int jj = tid >> 4;         // 0..15
    const f32x4 us4 = *(const f32x4*)(us + h);
    const f32x4 uo4 = *(const f32x4*)(uo + h);
    const f32x4 cv4 = *(const f32x4*)(cv + h);
    f32x4 base;
    base.x = cv4.x + dsi * us4.x;
    base.y = cv4.y + dsi * us4.y;
    base.z = cv4.z + dsi * us4.z;
    base.w = cv4.w + dsi * us4.w;

#pragma unroll
    for (int s = 0; s < 4; ++s) {
        int j = j0 + s * 16 + jj;
        float d = sd[s * 16 + jj];
        const float* arow;
        const float* garow;
        if (j == i) { arow = Bdiag + (size_t)i * HDIM; garow = Gh + (size_t)i * HDIM; }
        else        { arow = Acomb + (size_t)j * HDIM; garow = Ah + (size_t)j * HDIM; }
        f32x4 a4 = *(const f32x4*)(arow + h);
        f32x4 g4 = *(const f32x4*)(garow + h);
        f32x4 o;
        o.x = g4.x / (1.f + __expf(-(a4.x + base.x + d * uo4.x)));
        o.y = g4.y / (1.f + __expf(-(a4.y + base.y + d * uo4.y)));
        o.z = g4.z / (1.f + __expf(-(a4.z + base.z + d * uo4.z)));
        o.w = g4.w / (1.f + __expf(-(a4.w + base.w + d * uo4.w)));
        // Non-temporal: write-once 256 MB stream must not thrash L2.
        __builtin_nontemporal_store(o, (f32x4*)(out + ((size_t)i * N_PTS + j) * HDIM + h));
    }
}

extern "C" void kernel_launch(void* const* d_in, const int* in_sizes, int n_in,
                              void* d_out, int out_size, void* d_ws, size_t ws_size,
                              hipStream_t stream) {
    const float* Ah     = (const float*)d_in[0];  // action_hidden_state [N,H]
    const float* Gh     = (const float*)d_in[1];  // goal_hidden_state   [N,H]
    const float* goal   = (const float*)d_in[2];  // [N,2]
    const float* action = (const float*)d_in[3];  // [N,2]
    const float* wd     = (const float*)d_in[4];  // [E]
    const float* bd     = (const float*)d_in[5];  // [E]
    const float* Wg     = (const float*)d_in[6];  // [2E+2H, H] row-major
    const float* bg     = (const float*)d_in[7];  // [H]
    float* out = (float*)d_out;

    char* ws = (char*)d_ws;
    float* Acomb = (float*)ws;                                  // 256 KB
    float* Bdiag = (float*)(ws + (size_t)N_PTS * HDIM * 4);     // 256 KB
    float* us    = (float*)(ws + (size_t)2 * N_PTS * HDIM * 4); // 64 f
    float* uo    = us + HDIM;
    float* cvv   = us + 2 * HDIM;

    ig_precompute_vecs<<<1, 64, 0, stream>>>(wd, bd, Wg, bg, us, uo, cvv);
    ig_precompute_gemm<<<64, 128, 0, stream>>>(Ah, Gh, Wg, Acomb, Bdiag);
    ig_main<<<dim3(N_PTS / 64, N_PTS), 256, 0, stream>>>(Ah, Gh, goal, action,
                                                         Acomb, Bdiag, us, uo, cvv, out);
}
